// HybridMambaTransformer_3908420240342
// MI455X (gfx1250) — matrix operand order repaired and checked by reading, not compiled, not run on hardware
//
#include <hip/hip_runtime.h>
#include <hip/hip_bf16.h>

// ---- model dims ----
#define L_SEQ  1024
#define H_DIM  768
#define I_DIM  1536
#define NSTATE 16
#define R_DIM  48
#define NHEAD  12
#define HD_DIM 64
#define VOCAB  50257
#define VOCABP 50264   // vocab padded to multiple of 8 for 16B-aligned bf16 rows

typedef __attribute__((ext_vector_type(16))) __bf16 v16bf;
typedef __attribute__((ext_vector_type(8)))  __bf16 v8bfh;
typedef __attribute__((ext_vector_type(4)))  __bf16 v4bf;
typedef __attribute__((ext_vector_type(8)))  float  v8f;

__device__ __forceinline__ __bf16 f2bf(float f) {
  unsigned u = __builtin_bit_cast(unsigned, f);
  unsigned r = u + 0x7FFFu + ((u >> 16) & 1u);     // round-to-nearest-even
  unsigned short h = (unsigned short)(r >> 16);
  return __builtin_bit_cast(__bf16, h);
}
__device__ __forceinline__ float gelu_new_f(float x) {
  float c = x + 0.044715f * x * x * x;
  return 0.5f * x * (1.f + tanhf(0.7978845608028654f * c));
}
__device__ __forceinline__ float softplus_f(float x) {
  return (x > 20.f) ? x : log1pf(__expf(x));
}
__device__ __forceinline__ float silu_f(float x) {
  return x / (1.f + __expf(-x));
}

// k-column permutation so each lane's A fragment is one contiguous 32B read:
// physical [m][0..15]  = k {0..7, 16..23}   (hl=0 fragment)
// physical [m][16..31] = k {8..15, 24..31}  (hl=1 fragment)
__device__ __forceinline__ int kperm(int c) {
  int g = c >> 3;
  return ((g & 1) << 4) | ((g >> 1) << 3) | (c & 7);
}

// ======================================================================
// Tiled bf16-WMMA GEMM: C = act(alpha * A@B(^T) + bias) (+Res)
// 128 threads = 4 waves (2x2); block tile 64x64; each wave 32x32
// (2 A-frags x 2 B-frags -> 4 v_wmma_f32_16x16x32_bf16 per K-step).
// A: f32 (converted during LDS staging). B: f32 or pre-converted bf16 (bQ).
// ======================================================================
__global__ __launch_bounds__(128)
void wmma_gemm_kernel(const float* __restrict__ A, const void* __restrict__ Bv,
                      const float* __restrict__ bias, const float* __restrict__ Res,
                      float* __restrict__ C,
                      int M, int N, int Kd, int lda, int ldb, int ldc,
                      long long sA, long long sB, long long sC,
                      int bQ, int transB, int act, float alpha, int causal)
{
  __shared__ alignas(32) __bf16 As[64][32];   // [m][perm(k)]
  __shared__ alignas(32) __bf16 Bs[64][32];   // [n][k]

  const int tid  = threadIdx.x;
  const int lane = tid & 31;
  const int wave = tid >> 5;
  const int wm = wave >> 1, wn = wave & 1;    // 2x2 wave grid
  const int hl = lane >> 4, nl = lane & 15;
  const int bm = blockIdx.y * 64, bn = blockIdx.x * 64;
  const int bz = blockIdx.z;

  const float*  Ab  = A + (long long)bz * sA;
  const float*  Bbf = (const float*)Bv  + (long long)bz * sB;
  const __bf16* Bbh = (const __bf16*)Bv + (long long)bz * sB;
  float*        Cb  = C + (long long)bz * sC;
  const float*  Rb  = Res ? (Res + (long long)bz * sC) : nullptr;

  const bool mFull = (bm + 64 <= M);
  const bool nFull = (bn + 64 <= N);
  const bool ldaV  = ((lda & 3) == 0);
  const bool ldbV4 = ((ldb & 3) == 0);
  const bool ldbV8 = ((ldb & 7) == 0);

  v8f acc[2][2] = {};

  for (int k0 = 0; k0 < Kd; k0 += 32) {
    const bool kFull = (k0 + 32 <= Kd);
    // ---------------- stage A (64 x 32) ----------------
    if (mFull && kFull && ldaV) {
      for (int e = tid; e < 64 * 8; e += 128) {
        int r = e >> 3, c4 = e & 7;
        float4 v = *(const float4*)&Ab[(long long)(bm + r) * lda + k0 + c4 * 4];
        int g = c4 >> 1;
        int p4 = ((g & 1) << 2) | ((g >> 1) << 1) | (c4 & 1);
        v4bf pk = { f2bf(v.x), f2bf(v.y), f2bf(v.z), f2bf(v.w) };
        *(v4bf*)&As[r][p4 * 4] = pk;
      }
    } else {
      for (int e = tid; e < 64 * 32; e += 128) {
        int r = e >> 5, c = e & 31;
        float v = 0.f;
        int gr = bm + r, gc = k0 + c;
        if (gr < M && gc < Kd) v = Ab[(long long)gr * lda + gc];
        As[r][kperm(c)] = f2bf(v);
      }
    }
    // ---------------- stage B (32k x 64n, stored n-major) ----------------
    if (bQ) {  // pre-converted bf16 weights, row-major [k][n]
      if (nFull && kFull && ldbV8) {
        for (int e = tid; e < 32 * 8; e += 128) {
          int k = e >> 3, c8 = e & 7;
          v8bfh v = *(const v8bfh*)&Bbh[(long long)(k0 + k) * ldb + bn + c8 * 8];
          #pragma unroll
          for (int j = 0; j < 8; ++j) Bs[c8 * 8 + j][k] = v[j];
        }
      } else {
        const __bf16 zb = __builtin_bit_cast(__bf16, (unsigned short)0);
        for (int e = tid; e < 32 * 64; e += 128) {
          int k = e >> 6, n = e & 63;
          int gk = k0 + k, gn = bn + n;
          Bs[n][k] = (gk < Kd && gn < N) ? Bbh[(long long)gk * ldb + gn] : zb;
        }
      }
    } else if (nFull && kFull) {
      if (transB && ldbV4) {
        for (int e = tid; e < 64 * 8; e += 128) {
          int n = e >> 3, c4 = e & 7;
          float4 v = *(const float4*)&Bbf[(long long)(bn + n) * ldb + k0 + c4 * 4];
          v4bf pk = { f2bf(v.x), f2bf(v.y), f2bf(v.z), f2bf(v.w) };
          *(v4bf*)&Bs[n][c4 * 4] = pk;
        }
      } else if (!transB && ldbV4) {
        for (int e = tid; e < 32 * 16; e += 128) {
          int k = e >> 4, c4 = e & 15;
          float4 v = *(const float4*)&Bbf[(long long)(k0 + k) * ldb + bn + c4 * 4];
          Bs[c4 * 4 + 0][k] = f2bf(v.x);
          Bs[c4 * 4 + 1][k] = f2bf(v.y);
          Bs[c4 * 4 + 2][k] = f2bf(v.z);
          Bs[c4 * 4 + 3][k] = f2bf(v.w);
        }
      } else {  // full tile, unaligned ldb: branch-free scalar
        for (int e = tid; e < 32 * 64; e += 128) {
          int k = e >> 6, n = e & 63;
          float v = transB ? Bbf[(long long)(bn + n) * ldb + k0 + k]
                           : Bbf[(long long)(k0 + k) * ldb + bn + n];
          Bs[n][k] = f2bf(v);
        }
      }
    } else {
      for (int e = tid; e < 32 * 64; e += 128) {
        int k = e >> 6, n = e & 63;
        float v = 0.f;
        int gk = k0 + k, gn = bn + n;
        if (gk < Kd && gn < N)
          v = transB ? Bbf[(long long)gn * ldb + gk]
                     : Bbf[(long long)gk * ldb + gn];
        Bs[n][k] = f2bf(v);
      }
    }
    // prefetch next K tile into L2 while we compute on this one
    if (k0 + 32 < Kd) {
      __builtin_prefetch(&Ab[(long long)(bm + (tid & 63)) * lda + k0 + 32], 0, 1);
      if (bQ)
        __builtin_prefetch(&Bbh[(long long)(k0 + 32 + (tid & 31)) * ldb + bn], 0, 1);
      else if (transB)
        __builtin_prefetch(&Bbf[(long long)(bn + (tid & 63)) * ldb + k0 + 32], 0, 1);
      else
        __builtin_prefetch(&Bbf[(long long)(k0 + 32 + (tid & 31)) * ldb + bn], 0, 1);
    }
    __syncthreads();

    // ---------------- 4 WMMAs per wave ----------------
    const int m0 = wm * 32 + nl, m1 = m0 + 16;
    const int n0 = wn * 32 + nl, n1 = n0 + 16;
    v16bf a0 = *(const v16bf*)&As[m0][hl * 16];
    v16bf a1 = *(const v16bf*)&As[m1][hl * 16];
    v16bf b0 = __builtin_shufflevector(*(const v8bfh*)&Bs[n0][hl * 8], *(const v8bfh*)&Bs[n0][16 + hl * 8], 0,1,2,3,4,5,6,7,8,9,10,11,12,13,14,15);
    v16bf b1 = __builtin_shufflevector(*(const v8bfh*)&Bs[n1][hl * 8], *(const v8bfh*)&Bs[n1][16 + hl * 8], 0,1,2,3,4,5,6,7,8,9,10,11,12,13,14,15);

    acc[0][0] = __builtin_amdgcn_wmma_f32_16x16x32_bf16(false, a0, false, b0, (short)0, acc[0][0], false, false);
    acc[0][1] = __builtin_amdgcn_wmma_f32_16x16x32_bf16(false, a0, false, b1, (short)0, acc[0][1], false, false);
    acc[1][0] = __builtin_amdgcn_wmma_f32_16x16x32_bf16(false, a1, false, b0, (short)0, acc[1][0], false, false);
    acc[1][1] = __builtin_amdgcn_wmma_f32_16x16x32_bf16(false, a1, false, b1, (short)0, acc[1][1], false, false);
    __syncthreads();
  }

  // epilogue: C/D layout -- lane: col = nl; VGPR v: row = v + 8*hl
  #pragma unroll
  for (int sm = 0; sm < 2; ++sm)
  #pragma unroll
  for (int sn = 0; sn < 2; ++sn)
  #pragma unroll
  for (int v = 0; v < 8; ++v) {
    int row = bm + wm * 32 + sm * 16 + v + 8 * hl;
    int col = bn + wn * 32 + sn * 16 + nl;
    if (row < M && col < N) {
      float val = acc[sm][sn][v] * alpha;
      if (bias) val += bias[col];
      if (act == 1) val = gelu_new_f(val);
      else if (act == 2) val = softplus_f(val);
      if (causal && col > row) val = -1e30f;
      if (Rb) val += Rb[(long long)row * ldc + col];
      Cb[(long long)row * ldc + col] = val;
    }
  }
}

// ======================================================================
// weight f32 -> bf16 pre-conversion (row pitch padded for aligned loads)
// ======================================================================
__global__ __launch_bounds__(256)
void cvt_bf16_kernel(const float* __restrict__ src, __bf16* __restrict__ dst,
                     int cols, int ldp)
{
  int r = blockIdx.y;
  int c = blockIdx.x * 256 + threadIdx.x;
  if (c < cols) dst[(long long)r * ldp + c] = f2bf(src[(long long)r * cols + c]);
}

// ======================================================================
// elementwise / reduction kernels
// ======================================================================
__global__ __launch_bounds__(256)
void embed_kernel(const int* __restrict__ ids, const float* __restrict__ eW,
                  const float* __restrict__ pW, float* __restrict__ X)
{
  int t = blockIdx.x;
  long long id = ids[t];
  for (int h = threadIdx.x; h < H_DIM; h += 256)
    X[(long long)t * H_DIM + h] = eW[id * H_DIM + h] + pW[(long long)t * H_DIM + h];
}

// rms==1: RMSNorm (no bias); rms==0: LayerNorm
__global__ __launch_bounds__(256)
void norm_kernel(const float* __restrict__ Xin, const float* __restrict__ w,
                 const float* __restrict__ b, float* __restrict__ Out,
                 int n, int rms)
{
  const long long row = blockIdx.x;
  const float* x = Xin + row * n;
  float*       o = Out + row * n;
  __shared__ float red[256];
  float mean = 0.f;
  if (!rms) {
    float s = 0.f;
    for (int j = threadIdx.x; j < n; j += 256) s += x[j];
    red[threadIdx.x] = s; __syncthreads();
    for (int off = 128; off > 0; off >>= 1) {
      if (threadIdx.x < off) red[threadIdx.x] += red[threadIdx.x + off];
      __syncthreads();
    }
    mean = red[0] / n;
    __syncthreads();
  }
  float v = 0.f;
  for (int j = threadIdx.x; j < n; j += 256) { float d = x[j] - mean; v += d * d; }
  red[threadIdx.x] = v; __syncthreads();
  for (int off = 128; off > 0; off >>= 1) {
    if (threadIdx.x < off) red[threadIdx.x] += red[threadIdx.x + off];
    __syncthreads();
  }
  float rstd = rsqrtf(red[0] / n + 1e-5f);
  for (int j = threadIdx.x; j < n; j += 256) {
    float y = (x[j] - mean) * rstd * w[j];
    if (!rms) y += b[j];
    o[j] = y;
  }
}

// depthwise causal conv (K=4) + SiLU ; input = T1[:, :I] (ld 2I), out LxI
__global__ __launch_bounds__(256)
void conv_silu_kernel(const float* __restrict__ hsIn, const float* __restrict__ Wc,
                      const float* __restrict__ bc, float* __restrict__ Out)
{
  int i = blockIdx.y * 256 + threadIdx.x;
  int t = blockIdx.x;
  float acc = bc[i];
  #pragma unroll
  for (int k = 0; k < 4; ++k) {
    int tt = t + k - 3;
    if (tt >= 0) acc += hsIn[(long long)tt * (2 * I_DIM) + i] * Wc[i * 4 + k];
  }
  Out[(long long)t * I_DIM + i] = silu_f(acc);
}

// sequential selective-scan: one channel per lane, 16-state in registers
__global__ __launch_bounds__(256)
void mamba_scan_kernel(const float* __restrict__ dt, const float* __restrict__ hs,
                       const float* __restrict__ ssm, const float* __restrict__ Alog,
                       const float* __restrict__ Dp, const float* __restrict__ gateBuf,
                       float* __restrict__ Yout)
{
  const int i = blockIdx.x * 256 + threadIdx.x;   // channel < 1536
  float st[NSTATE], Av[NSTATE];
  #pragma unroll
  for (int n = 0; n < NSTATE; ++n) {
    st[n] = 0.f;
    Av[n] = -__expf(Alog[(long long)i * NSTATE + n]);
  }
  const float Dv = Dp[i];
  __shared__ float Bt[NSTATE], Ct[NSTATE];
  for (int t = 0; t < L_SEQ; ++t) {
    if (threadIdx.x < 32) {
      float v = ssm[(long long)t * 80 + 48 + threadIdx.x];
      if (threadIdx.x < 16) Bt[threadIdx.x] = v; else Ct[threadIdx.x - 16] = v;
    }
    __syncthreads();
    float d  = dt[(long long)t * I_DIM + i];
    float u  = hs[(long long)t * I_DIM + i];
    float du = d * u;
    float y = 0.f;
    #pragma unroll
    for (int n = 0; n < NSTATE; ++n) {
      float dA = __expf(d * Av[n]);
      st[n] = dA * st[n] + du * Bt[n];
      y += st[n] * Ct[n];
    }
    y += u * Dv;
    float g = gateBuf[(long long)t * (2 * I_DIM) + I_DIM + i];
    y *= silu_f(g);
    Yout[(long long)t * I_DIM + i] = y;
    __syncthreads();
  }
}

__global__ __launch_bounds__(256)
void softmax_kernel(float* __restrict__ S, int n)
{
  float* row = S + (long long)blockIdx.x * n;
  __shared__ float red[256];
  float m = -1e30f;
  for (int j = threadIdx.x; j < n; j += 256) m = fmaxf(m, row[j]);
  red[threadIdx.x] = m; __syncthreads();
  for (int off = 128; off > 0; off >>= 1) {
    if (threadIdx.x < off) red[threadIdx.x] = fmaxf(red[threadIdx.x], red[threadIdx.x + off]);
    __syncthreads();
  }
  m = red[0]; __syncthreads();
  float s = 0.f;
  for (int j = threadIdx.x; j < n; j += 256) {
    float e = __expf(row[j] - m);
    row[j] = e; s += e;
  }
  red[threadIdx.x] = s; __syncthreads();
  for (int off = 128; off > 0; off >>= 1) {
    if (threadIdx.x < off) red[threadIdx.x] += red[threadIdx.x + off];
    __syncthreads();
  }
  float inv = 1.f / red[0];
  for (int j = threadIdx.x; j < n; j += 256) row[j] *= inv;
}

// ======================================================================
// host side
// ======================================================================
static void launch_gemm(hipStream_t stream, const float* A, const void* B, int bQ,
                        const float* bias, const float* Res, float* C,
                        int M, int N, int Kd, int lda, int ldb, int ldc,
                        long long sA, long long sB, long long sC, int batch,
                        int transB, int act, float alpha, int causal)
{
  dim3 g((unsigned)((N + 63) / 64), (unsigned)((M + 63) / 64), (unsigned)batch);
  wmma_gemm_kernel<<<g, 128, 0, stream>>>(A, B, bias, Res, C, M, N, Kd,
                                          lda, ldb, ldc, sA, sB, sC,
                                          bQ, transB, act, alpha, causal);
}

extern "C" void kernel_launch(void* const* d_in, const int* in_sizes, int n_in,
                              void* d_out, int out_size, void* d_ws, size_t ws_size,
                              hipStream_t stream) {
  (void)in_sizes; (void)n_in; (void)out_size; (void)ws_size;
  const int*   ids      = (const int*)  d_in[0];
  const float* embed_W  = (const float*)d_in[1];
  const float* pos_W    = (const float*)d_in[2];
  const float* m_norm_w = (const float*)d_in[3];
  const float* m_in_proj= (const float*)d_in[4];
  const float* m_conv_w = (const float*)d_in[5];
  const float* m_conv_b = (const float*)d_in[6];
  const float* m_x_proj = (const float*)d_in[7];
  const float* m_dt_w   = (const float*)d_in[8];
  const float* m_dt_b   = (const float*)d_in[9];
  const float* m_A_log  = (const float*)d_in[10];
  const float* m_D      = (const float*)d_in[11];
  const float* m_out_proj=(const float*)d_in[12];
  const float* g_ln1_w  = (const float*)d_in[13];
  const float* g_ln1_b  = (const float*)d_in[14];
  const float* g_attn_w = (const float*)d_in[15];
  const float* g_attn_b = (const float*)d_in[16];
  const float* g_aproj_w= (const float*)d_in[17];
  const float* g_aproj_b= (const float*)d_in[18];
  const float* g_ln2_w  = (const float*)d_in[19];
  const float* g_ln2_b  = (const float*)d_in[20];
  const float* g_fc_w   = (const float*)d_in[21];
  const float* g_fc_b   = (const float*)d_in[22];
  const float* g_proj_w = (const float*)d_in[23];
  const float* g_proj_b = (const float*)d_in[24];
  const float* lnf_w    = (const float*)d_in[25];
  const float* lnf_b    = (const float*)d_in[26];
  const float* head_W   = (const float*)d_in[27];
  float* out = (float*)d_out;

  // workspace layout (floats)
  float* ws  = (float*)d_ws;
  float* X   = ws;                                  // L x H
  float* Hn  = X   + (long long)L_SEQ * H_DIM;      // L x H (norm out)
  float* T1  = Hn  + (long long)L_SEQ * H_DIM;      // L x 3072
  float* T2  = T1  + (long long)L_SEQ * 3072;       // L x 3072 / L x I
  float* Yb  = T2  + (long long)L_SEQ * 3072;       // L x I
  float* SSM = Yb  + (long long)L_SEQ * I_DIM;      // L x 80
  float* DT  = SSM + (long long)L_SEQ * 80;         // L x I
  float* CTX = DT  + (long long)L_SEQ * I_DIM;      // L x H
  float* SC  = CTX + (long long)L_SEQ * H_DIM;      // NH x L x L

  // bf16 weight arena (after SC)
  __bf16* WB = (__bf16*)(SC + (long long)NHEAD * L_SEQ * L_SEQ);
  long long wb_off = 0;
  auto cvt = [&](const float* src, int rows, int cols, int ldp) -> const __bf16* {
    wb_off = (wb_off + 7) & ~7LL;                 // 16B-aligned allocations
    __bf16* dst = WB + wb_off;
    wb_off += (long long)rows * ldp;
    dim3 g((unsigned)((cols + 255) / 256), (unsigned)rows);
    cvt_bf16_kernel<<<g, 256, 0, stream>>>(src, dst, cols, ldp);
    return dst;
  };

  // pre-convert all weight matrices to bf16 (halves HBM weight traffic)
  const __bf16 *W_in[9], *W_x[9], *W_dt[9], *W_out[9];
  for (int m = 0; m < 9; ++m) {
    W_in[m]  = cvt(m_in_proj  + (long long)m * H_DIM * 2 * I_DIM, H_DIM, 2 * I_DIM, 2 * I_DIM);
    W_x[m]   = cvt(m_x_proj   + (long long)m * I_DIM * 80,        I_DIM, 80,        80);
    W_dt[m]  = cvt(m_dt_w     + (long long)m * R_DIM * I_DIM,     R_DIM, I_DIM,     I_DIM);
    W_out[m] = cvt(m_out_proj + (long long)m * I_DIM * H_DIM,     I_DIM, H_DIM,     H_DIM);
  }
  const __bf16 *W_qkv[3], *W_ap[3], *W_fc[3], *W_pr[3];
  for (int a = 0; a < 3; ++a) {
    W_qkv[a] = cvt(g_attn_w  + (long long)a * H_DIM * 3 * H_DIM, H_DIM, 3 * H_DIM, 3 * H_DIM);
    W_ap[a]  = cvt(g_aproj_w + (long long)a * H_DIM * H_DIM,     H_DIM, H_DIM,     H_DIM);
    W_fc[a]  = cvt(g_fc_w    + (long long)a * H_DIM * 4 * H_DIM, H_DIM, 4 * H_DIM, 4 * H_DIM);
    W_pr[a]  = cvt(g_proj_w  + (long long)a * 4 * H_DIM * H_DIM, 4 * H_DIM, H_DIM, H_DIM);
  }
  const __bf16* W_head = cvt(head_W, H_DIM, VOCAB, VOCABP);

  // embedding
  embed_kernel<<<L_SEQ, 256, 0, stream>>>(ids, embed_W, pos_W, X);

  int mi = 0, ai = 0;
  for (int layer = 0; layer < 12; ++layer) {
    bool is_attn = (layer == 3) || (layer == 7) || (layer == 11);
    if (is_attn) {
      // LN1
      norm_kernel<<<L_SEQ, 256, 0, stream>>>(X, g_ln1_w + (long long)ai * H_DIM,
                                             g_ln1_b + (long long)ai * H_DIM, Hn, H_DIM, 0);
      // QKV
      launch_gemm(stream, Hn, W_qkv[ai], 1,
                  g_attn_b + (long long)ai * 3 * H_DIM, nullptr, T1,
                  L_SEQ, 3 * H_DIM, H_DIM, H_DIM, 3 * H_DIM, 3 * H_DIM,
                  0, 0, 0, 1, 0, 0, 1.0f, 0);
      // scores = Q K^T / 8, causal-masked (batched over heads)
      launch_gemm(stream, T1, T1 + H_DIM, 0, nullptr, nullptr, SC,
                  L_SEQ, L_SEQ, HD_DIM, 3 * H_DIM, 3 * H_DIM, L_SEQ,
                  HD_DIM, HD_DIM, (long long)L_SEQ * L_SEQ, NHEAD,
                  1 /*transB*/, 0, 0.125f, 1 /*causal*/);
      softmax_kernel<<<NHEAD * L_SEQ, 256, 0, stream>>>(SC, L_SEQ);
      // ctx = P V
      launch_gemm(stream, SC, T1 + 2 * H_DIM, 0, nullptr, nullptr, CTX,
                  L_SEQ, HD_DIM, L_SEQ, L_SEQ, 3 * H_DIM, H_DIM,
                  (long long)L_SEQ * L_SEQ, HD_DIM, HD_DIM, NHEAD,
                  0, 0, 1.0f, 0);
      // attn out proj + residual
      launch_gemm(stream, CTX, W_ap[ai], 1,
                  g_aproj_b + (long long)ai * H_DIM, X, X,
                  L_SEQ, H_DIM, H_DIM, H_DIM, H_DIM, H_DIM,
                  0, 0, 0, 1, 0, 0, 1.0f, 0);
      // LN2
      norm_kernel<<<L_SEQ, 256, 0, stream>>>(X, g_ln2_w + (long long)ai * H_DIM,
                                             g_ln2_b + (long long)ai * H_DIM, Hn, H_DIM, 0);
      // MLP fc (gelu) then proj (+residual)
      launch_gemm(stream, Hn, W_fc[ai], 1,
                  g_fc_b + (long long)ai * 4 * H_DIM, nullptr, T1,
                  L_SEQ, 4 * H_DIM, H_DIM, H_DIM, 4 * H_DIM, 4 * H_DIM,
                  0, 0, 0, 1, 0, 1 /*gelu*/, 1.0f, 0);
      launch_gemm(stream, T1, W_pr[ai], 1,
                  g_proj_b + (long long)ai * H_DIM, X, X,
                  L_SEQ, H_DIM, 4 * H_DIM, 4 * H_DIM, H_DIM, H_DIM,
                  0, 0, 0, 1, 0, 0, 1.0f, 0);
      ++ai;
    } else {
      // RMSNorm
      norm_kernel<<<L_SEQ, 256, 0, stream>>>(X, m_norm_w + (long long)mi * H_DIM,
                                             nullptr, Hn, H_DIM, 1);
      // in_proj: L x 2I  (hs | gate)
      launch_gemm(stream, Hn, W_in[mi], 1, nullptr, nullptr, T1,
                  L_SEQ, 2 * I_DIM, H_DIM, H_DIM, 2 * I_DIM, 2 * I_DIM,
                  0, 0, 0, 1, 0, 0, 1.0f, 0);
      // conv + silu -> T2 (L x I)
      conv_silu_kernel<<<dim3(L_SEQ, I_DIM / 256), 256, 0, stream>>>(
          T1, m_conv_w + (long long)mi * I_DIM * 4, m_conv_b + (long long)mi * I_DIM, T2);
      // x_proj: L x 80
      launch_gemm(stream, T2, W_x[mi], 1, nullptr, nullptr, SSM,
                  L_SEQ, 80, I_DIM, I_DIM, 80, 80,
                  0, 0, 0, 1, 0, 0, 1.0f, 0);
      // dt = softplus(ssm[:, :48] @ Wdt + bdt) : L x I
      launch_gemm(stream, SSM, W_dt[mi], 1,
                  m_dt_b + (long long)mi * I_DIM, nullptr, DT,
                  L_SEQ, I_DIM, R_DIM, 80, I_DIM, I_DIM,
                  0, 0, 0, 1, 0, 2 /*softplus*/, 1.0f, 0);
      // selective scan -> Yb
      mamba_scan_kernel<<<I_DIM / 256, 256, 0, stream>>>(
          DT, T2, SSM, m_A_log + (long long)mi * I_DIM * NSTATE,
          m_D + (long long)mi * I_DIM, T1, Yb);
      // out_proj + residual
      launch_gemm(stream, Yb, W_out[mi], 1, nullptr, X, X,
                  L_SEQ, H_DIM, I_DIM, I_DIM, H_DIM, H_DIM,
                  0, 0, 0, 1, 0, 0, 1.0f, 0);
      ++mi;
    }
  }

  // final LN + LM head (bf16 weights, padded pitch -> aligned fast path)
  norm_kernel<<<L_SEQ, 256, 0, stream>>>(X, lnf_w, lnf_b, Hn, H_DIM, 0);
  launch_gemm(stream, Hn, W_head, 1, nullptr, nullptr, out,
              L_SEQ, VOCAB, H_DIM, H_DIM, VOCABP, VOCAB,
              0, 0, 0, 1, 0, 0, 1.0f, 0);
}
